// RecursiveBlock_51934744543660
// MI455X (gfx1250) — matrix operand order repaired and checked by reading, not compiled, not run on hardware
//
#include <hip/hip_runtime.h>

// ---------------------------------------------------------------------------
// Shapes
//   x      : [8, 512, 96, 96] fp32
//   w_mix  : [2, 256, 256]    fp32   (grouped 1x1 conv, groups=2)
//   w_dw   : [512, 1, 3, 3]   fp32   (depthwise 3x3, pad 1)
//   gammas : [4, 1, 512, 1, 1]
//   betas  : [4, 1, 512, 1, 1]
//   alpha  : scalar
// ---------------------------------------------------------------------------

#define BATCH 8
#define CCH   512
#define HH    96
#define WW    96
#define HWS   (HH*WW)          // 9216 spatial
#define GRP   2
#define CG    256              // channels per group (M and K of the GEMM)
#define NTILE 64               // spatial tile per workgroup
#define KBLK  64               // K staged per LDS round
#define PW    36               // LDS pitch in dwords (32 k-pairs + 4 pad)

typedef __attribute__((ext_vector_type(16))) __bf16 v16bf;
typedef __attribute__((ext_vector_type(8)))  float  v8f;
typedef __attribute__((ext_vector_type(4)))  int    v4i;

struct Frag32 { uint4 a, b; };   // 32 bytes == v16bf

static __device__ inline v16bf frag_cast(Frag32 f) {
    return __builtin_bit_cast(v16bf, f);
}

// fp32 -> bf16 bits, round-to-nearest-even (finite inputs)
static __device__ inline unsigned short f2bf_bits(float f) {
    unsigned u = __builtin_bit_cast(unsigned, f);
    unsigned r = u + 0x7FFFu + ((u >> 16) & 1u);
    return (unsigned short)(r >> 16);
}
static __device__ inline float bf2f(unsigned short h) {
    return __builtin_bit_cast(float, (unsigned)h << 16);
}

// ---------------------------------------------------------------------------
// Optional CDNA5 async global->LDS path for raw fp32 plane copies.
// Probe-learned signature: (v4i AS(1)* gsrc, v4i AS(3)* ldst, int off, int cpol)
// ---------------------------------------------------------------------------
#if defined(__has_builtin)
#  if __has_builtin(__builtin_amdgcn_global_load_async_to_lds_b128)
#    define ASYNC_LDS_LOAD 1
#  endif
#  if __has_builtin(__builtin_amdgcn_s_wait_asynccnt)
#    define HAVE_WAIT_ASYNC_BUILTIN 1
#  endif
#endif

typedef __attribute__((address_space(1))) v4i GlobV4;
typedef __attribute__((address_space(3))) v4i LdsV4;

static __device__ inline void plane_load(float* plane, const float* src, int tid) {
#ifdef ASYNC_LDS_LOAD
    #pragma unroll
    for (int i = 0; i < HWS / (256 * 4); ++i) {          // 9 iters of b128
        int e = (tid + i * 256) * 4;
        __builtin_amdgcn_global_load_async_to_lds_b128(
            (GlobV4*)(src + e), (LdsV4*)(plane + e), 0, 0);
    }
#  ifdef HAVE_WAIT_ASYNC_BUILTIN
    __builtin_amdgcn_s_wait_asynccnt(0);
#  else
    asm volatile("s_wait_asynccnt 0" ::: "memory");
#  endif
#else
    for (int i = tid; i < HWS; i += 256) plane[i] = src[i];
#endif
}

// ---------------------------------------------------------------------------
// Kernel 0: split w_mix into bf16 hi/lo, pre-swizzled into this file's A-fragment
// order for V_WMMA_F32_16X16X32_BF16 (contiguous k, same as B; not the ISA table order):
//   lane l (<16): row M = l,    k_local = e
//   lane l (>=16): row M = l-16, k_local = e + 16
// Fragment index: ((g*8 + ktile)*16 + mtile)*32 + lane, 16 bf16 each.
// ---------------------------------------------------------------------------
__global__ void prep_w_kernel(const float* __restrict__ wmix,
                              unsigned short* __restrict__ ahi,
                              unsigned short* __restrict__ alo) {
    int idx = blockIdx.x * blockDim.x + threadIdx.x;     // 0 .. 8191
    if (idx >= GRP * 8 * 16 * 32) return;
    int lane  = idx & 31;
    int t     = idx >> 5;
    int mtile = t & 15;  t >>= 4;
    int kt    = t & 7;
    int g     = t >> 3;
    int m     = 16 * mtile + (lane & 15);
    int base  = (g * CG + m) * CG;
    size_t fo = (size_t)idx * 16;
    #pragma unroll
    for (int e = 0; e < 16; ++e) {
        int klocal = e + ((lane >= 16) ? 16 : 0);
        int k = 32 * kt + klocal;
        float f = wmix[base + k];
        unsigned short hb = f2bf_bits(f);
        ahi[fo + e] = hb;
        alo[fo + e] = f2bf_bits(f - bf2f(hb));
    }
}

// ---------------------------------------------------------------------------
// Kernel 1: grouped 1x1 conv as GEMM  out[m][s] = sum_k W[g][m][k] * x[g,k][s]
// bf16 hi/lo split: W*X ~= Whi*Xhi + Whi*Xlo + Wlo*Xhi  (fp32 accumulate).
// Activations converted to bf16 ONCE at staging time and stored in LDS
// pre-packed as K-pair dwords in exact B-fragment order, so the inner loop is
// pure ds_load_b128 + v_wmma (no conversion VALU, no redundancy across waves).
// Channel-shuffle (cout = 2*m + g) folded into the store.
// ---------------------------------------------------------------------------
__global__ __launch_bounds__(256)
void gemm_mix_kernel(const float* __restrict__ x,
                     const unsigned short* __restrict__ ahi,
                     const unsigned short* __restrict__ alo,
                     float* __restrict__ hmid) {
    __shared__ unsigned xhi[NTILE * PW];   // [s][kpair] bf16-hi pairs
    __shared__ unsigned xlo[NTILE * PW];   // [s][kpair] bf16-lo pairs

    int blk   = blockIdx.x;
    int stile = blk % (HWS / NTILE);     // 144 spatial tiles
    int bg    = blk / (HWS / NTILE);
    int g     = bg & 1;
    int b     = bg >> 1;
    int s0    = stile * NTILE;

    int tid  = threadIdx.x;
    int lane = tid & 31;
    int wv   = tid >> 5;

    const float* xbase = x + ((size_t)(b * CCH + g * CG) * HWS) + s0;

    v8f acc[2][4];
    #pragma unroll
    for (int i = 0; i < 2; ++i)
        #pragma unroll
        for (int j = 0; j < 4; ++j)
            #pragma unroll
            for (int r = 0; r < 8; ++r) acc[i][j][r] = 0.0f;

    int kp  = tid >> 3;      // k-pair 0..31 staged by this thread
    int sq  = tid & 7;
    int s0t = sq * 8;        // 8 consecutive s per thread

    for (int kb = 0; kb < CG / KBLK; ++kb) {
        // ---- stage: two adjacent K rows, 8 s each; convert + pack pairs ----
        const float* r0 = xbase + (size_t)(kb * KBLK + 2 * kp) * HWS + s0t;
        const float* r1 = r0 + HWS;
        float4 a0 = *(const float4*)(r0);
        float4 a1 = *(const float4*)(r0 + 4);
        float4 b0 = *(const float4*)(r1);
        float4 b1 = *(const float4*)(r1 + 4);
        if (kb < CG / KBLK - 1) {              // prefetch next K-block lines
            __builtin_prefetch(r0 + (size_t)KBLK * HWS, 0, 0);
            __builtin_prefetch(r1 + (size_t)KBLK * HWS, 0, 0);
        }
        float f0[8] = { a0.x, a0.y, a0.z, a0.w, a1.x, a1.y, a1.z, a1.w };
        float f1[8] = { b0.x, b0.y, b0.z, b0.w, b1.x, b1.y, b1.z, b1.w };
        #pragma unroll
        for (int j = 0; j < 8; ++j) {
            unsigned short h0 = f2bf_bits(f0[j]);
            unsigned short h1 = f2bf_bits(f1[j]);
            unsigned short l0 = f2bf_bits(f0[j] - bf2f(h0));
            unsigned short l1 = f2bf_bits(f1[j] - bf2f(h1));
            xhi[(s0t + j) * PW + kp] = (unsigned)h0 | ((unsigned)h1 << 16);
            xlo[(s0t + j) * PW + kp] = (unsigned)l0 | ((unsigned)l1 << 16);
        }
        __syncthreads();

        #pragma unroll
        for (int kk = 0; kk < 2; ++kk) {            // two K=32 steps per block
            int kt = kb * 2 + kk;
            v16bf a_hi[2], a_lo[2];
            #pragma unroll
            for (int mt = 0; mt < 2; ++mt) {
                int mtile = 2 * wv + mt;
                size_t fo = ((size_t)((g * 8 + kt) * 16 + mtile) * 32 + lane) * 16;
                a_hi[mt] = frag_cast(*(const Frag32*)(ahi + fo));
                a_lo[mt] = frag_cast(*(const Frag32*)(alo + fo));
            }
            int ncol = lane & 15;
            int kpb  = kk * 16 + ((lane >= 16) ? 8 : 0);   // k-pair base
            #pragma unroll
            for (int nt = 0; nt < 4; ++nt) {
                int row = (nt * 16 + ncol) * PW + kpb;
                const uint4* hp = (const uint4*)&xhi[row];
                const uint4* lp = (const uint4*)&xlo[row];
                Frag32 fh; fh.a = hp[0]; fh.b = hp[1];
                Frag32 fl; fl.a = lp[0]; fl.b = lp[1];
                v16bf b_hi = frag_cast(fh);
                v16bf b_lo = frag_cast(fl);
                #pragma unroll
                for (int mt = 0; mt < 2; ++mt) {
                    acc[mt][nt] = __builtin_amdgcn_wmma_f32_16x16x32_bf16(
                        false, a_hi[mt], false, b_hi, (short)0, acc[mt][nt], false, false);
                    acc[mt][nt] = __builtin_amdgcn_wmma_f32_16x16x32_bf16(
                        false, a_hi[mt], false, b_lo, (short)0, acc[mt][nt], false, false);
                    acc[mt][nt] = __builtin_amdgcn_wmma_f32_16x16x32_bf16(
                        false, a_lo[mt], false, b_hi, (short)0, acc[mt][nt], false, false);
                }
            }
        }
        __syncthreads();
    }

    // ---- store D with channel shuffle: cout = 2*m + g ----
    int ncol = lane & 15;
    int radd = (lane >= 16) ? 8 : 0;
    #pragma unroll
    for (int mt = 0; mt < 2; ++mt) {
        int m0 = 32 * wv + 16 * mt;
        #pragma unroll
        for (int nt = 0; nt < 4; ++nt) {
            int s = s0 + nt * 16 + ncol;
            #pragma unroll
            for (int r = 0; r < 8; ++r) {
                int m    = m0 + r + radd;
                int cout = 2 * m + g;
                hmid[(size_t)(b * CCH + cout) * HWS + s] = acc[mt][nt][r];
            }
        }
    }
}

// ---------------------------------------------------------------------------
// Kernel 2: depthwise 3x3 (pad 1) IN-PLACE on hmid + fused instance-norm
// stats (mean, rstd). One block per (b,c) plane, plane cached in LDS.
// ---------------------------------------------------------------------------
__global__ __launch_bounds__(256)
void dw_stats_kernel(float* __restrict__ hbuf,
                     const float* __restrict__ wdw,
                     float* __restrict__ stats) {
    __shared__ float plane[HWS];
    __shared__ float red[512];
    int bc  = blockIdx.x;            // b*512 + c
    int c   = bc & (CCH - 1);
    int tid = threadIdx.x;
    float* hp = hbuf + (size_t)bc * HWS;

    plane_load(plane, hp, tid);
    float wk[9];
    #pragma unroll
    for (int j = 0; j < 9; ++j) wk[j] = wdw[c * 9 + j];
    __syncthreads();

    float sum = 0.f, ss = 0.f;
    float outv[36];
    #pragma unroll
    for (int i = 0; i < 36; ++i) {
        int idx = tid + i * 256;
        int y = idx / WW, xw = idx % WW;
        float a = 0.f;
        #pragma unroll
        for (int dy = -1; dy <= 1; ++dy) {
            int yy = y + dy;
            if (yy < 0 || yy > HH - 1) continue;
            #pragma unroll
            for (int dx = -1; dx <= 1; ++dx) {
                int xx = xw + dx;
                if (xx < 0 || xx > WW - 1) continue;
                a += plane[yy * WW + xx] * wk[(dy + 1) * 3 + (dx + 1)];
            }
        }
        outv[i] = a;
        sum += a;
        ss  += a * a;
    }
    #pragma unroll
    for (int i = 0; i < 36; ++i) hp[tid + i * 256] = outv[i];  // LDS holds source

    red[tid] = sum; red[256 + tid] = ss;
    __syncthreads();
    for (int off = 128; off > 0; off >>= 1) {
        if (tid < off) { red[tid] += red[tid + off]; red[256 + tid] += red[256 + tid + off]; }
        __syncthreads();
    }
    if (tid == 0) {
        float mean = red[0] * (1.f / (float)HWS);
        float var  = red[256] * (1.f / (float)HWS) - mean * mean;
        stats[2 * bc]     = mean;
        stats[2 * bc + 1] = rsqrtf(var + 1e-5f);
    }
}

// ---------------------------------------------------------------------------
// Kernel 3: entropy gate (circular rolls from LDS) + IN affine + exact GELU
// + residual combine:  x_new = x + gate*h + alpha*(x - x_prev)
// ---------------------------------------------------------------------------
__global__ __launch_bounds__(256)
void finalize_kernel(const float* __restrict__ xc, const float* __restrict__ xp,
                     const float* __restrict__ hdw, const float* __restrict__ stats,
                     const float* __restrict__ gammas, const float* __restrict__ betas,
                     const float* __restrict__ alphap, float* __restrict__ xo, int t) {
    __shared__ float plane[HWS];
    int bc  = blockIdx.x;
    int c   = bc & (CCH - 1);
    int tid = threadIdx.x;
    plane_load(plane, xc + (size_t)bc * HWS, tid);
    float mean  = stats[2 * bc];
    float rstd  = stats[2 * bc + 1];
    float gamma = gammas[t * CCH + c];
    float beta  = betas[t * CCH + c];
    float alpha = alphap[0];
    __syncthreads();

    #pragma unroll
    for (int i = 0; i < 36; ++i) {
        int idx = tid + i * 256;
        int y = idx / WW, xw = idx % WW;
        float v = plane[idx];
        float gsum = 0.f;
        #pragma unroll
        for (int dd = 0; dd < 3; ++dd) {
            int d  = 1 << dd;                 // 1, 2, 4
            int ym = y - d;  if (ym < 0) ym += HH;
            int xm = xw - d; if (xm < 0) xm += WW;
            gsum += fabsf(v - plane[ym * WW + xw]) + fabsf(v - plane[y * WW + xm]);
        }
        float gate = 1.f / (1.f + expf(-gsum * (1.f / 3.f)));
        float h  = (hdw[(size_t)bc * HWS + idx] - mean) * rstd * gamma + beta;
        float ge = 0.5f * h * (1.f + erff(h * 0.70710678118654752f));
        float pv = xp[(size_t)bc * HWS + idx];
        xo[(size_t)bc * HWS + idx] = v + gate * ge + alpha * (v - pv);
    }
}

// ---------------------------------------------------------------------------
extern "C" void kernel_launch(void* const* d_in, const int* in_sizes, int n_in,
                              void* d_out, int out_size, void* d_ws, size_t ws_size,
                              hipStream_t stream) {
    const float* x    = (const float*)d_in[0];
    const float* wmix = (const float*)d_in[1];
    const float* wdw  = (const float*)d_in[2];
    const float* gam  = (const float*)d_in[3];
    const float* bet  = (const float*)d_in[4];
    const float* alp  = (const float*)d_in[5];

    const size_t N = (size_t)BATCH * CCH * HWS;     // 37,748,736
    float* X0    = (float*)d_ws;
    float* X1    = X0 + N;
    float* Hbuf  = X1 + N;
    float* stats = Hbuf + N;                         // 4096 * 2 floats
    unsigned short* Ahi = (unsigned short*)(stats + 2 * BATCH * CCH);
    unsigned short* Alo = Ahi + GRP * 8 * 16 * 32 * 16;   // 131072 each
    float* out = (float*)d_out;

    prep_w_kernel<<<32, 256, 0, stream>>>(wmix, Ahi, Alo);

    const int gemm_grid  = BATCH * GRP * (HWS / NTILE);   // 2304
    const int plane_grid = BATCH * CCH;                   // 4096

    const float* cur  = x;
    const float* prev = x;
    float* bufs[4] = { X0, X1, X0, out };   // t=2 overwrites its prev (pointwise-safe)
    for (int t = 0; t < 4; ++t) {
        gemm_mix_kernel<<<gemm_grid, 256, 0, stream>>>(cur, Ahi, Alo, Hbuf);
        dw_stats_kernel<<<plane_grid, 256, 0, stream>>>(Hbuf, wdw, stats);
        finalize_kernel<<<plane_grid, 256, 0, stream>>>(cur, prev, Hbuf, stats,
                                                        gam, bet, alp, bufs[t], t);
        prev = cur;
        cur  = bufs[t];
    }
}
